// Attention_49091476194121
// MI455X (gfx1250) — matrix operand order repaired and checked by reading, not compiled, not run on hardware
//
#include <hip/hip_runtime.h>

#define DM   1024
#define SEQ  2048
#define BAT  4

typedef __attribute__((ext_vector_type(16))) __bf16   v16bf;
typedef __attribute__((ext_vector_type(8)))  float    v8f;
typedef __attribute__((ext_vector_type(4)))  unsigned u32x4;

union BF16Frag {
    v16bf    v;
    unsigned u[8];
    u32x4    q[2];
    __bf16   h[16];
};

__device__ __forceinline__ unsigned short f32_bf16(float f) {
    unsigned u = __builtin_bit_cast(unsigned, f);
    u += 0x7FFFu + ((u >> 16) & 1u);          // round-to-nearest-even
    return (unsigned short)(u >> 16);
}
__device__ __forceinline__ __bf16 f32_bf(float f) {
    return __builtin_bit_cast(__bf16, f32_bf16(f));
}
__device__ __forceinline__ unsigned pack2(float lo, float hi) {
    return (unsigned)f32_bf16(lo) | ((unsigned)f32_bf16(hi) << 16);
}

// A fragment (16x32, row = lane&15): dwords 0..3 at +kbA, 4..7 at +kbA+16
__device__ __forceinline__ v16bf load_fragA(const __bf16* p) {
    BF16Frag f;
    f.q[0] = *reinterpret_cast<const u32x4*>(p);
    f.q[1] = *reinterpret_cast<const u32x4*>(p + 16);
    return f.v;
}
// B fragment (32x16, col = lane&15, source transposed): dwords 0..3 at +kbB, 4..7 at +kbB+16
__device__ __forceinline__ v16bf load_fragB(const __bf16* p) {
    BF16Frag f;
    f.q[0] = *reinterpret_cast<const u32x4*>(p);
    f.q[1] = *reinterpret_cast<const u32x4*>(p + 16);
    return f.v;
}

struct Acc2x2 { v8f c[2][2]; };

// 32x32 output block per wave: C = A(MxK) * Bt(NxK)^T, all bf16, f32 accum.
__device__ __forceinline__ void gemm_tile_2x2(
    const __bf16* __restrict__ A,  int lda,
    const __bf16* __restrict__ Bt, int ldb,
    int row0, int col0, int Kdim, int lane, Acc2x2& acc)
{
    const int l15 = lane & 15;
    const int kbA = (lane < 16) ? 0 : 8;
    const int kbB = (lane < 16) ? 0 : 8;
    const __bf16* a0p = A  + (size_t)(row0 + l15)      * lda + kbA;
    const __bf16* a1p = A  + (size_t)(row0 + 16 + l15) * lda + kbA;
    const __bf16* b0p = Bt + (size_t)(col0 + l15)      * ldb + kbB;
    const __bf16* b1p = Bt + (size_t)(col0 + 16 + l15) * ldb + kbB;

    for (int k0 = 0; k0 < Kdim; k0 += 32) {
        if (k0 + 64 < Kdim) {                  // gfx1250 global_prefetch_b8
            __builtin_prefetch(a0p + k0 + 64, 0, 1);
            __builtin_prefetch(b0p + k0 + 64, 0, 1);
        }
        v16bf a0 = load_fragA(a0p + k0);
        v16bf a1 = load_fragA(a1p + k0);
        v16bf b0 = load_fragB(b0p + k0);
        v16bf b1 = load_fragB(b1p + k0);
        acc.c[0][0] = __builtin_amdgcn_wmma_f32_16x16x32_bf16(
                          false, a0, false, b0, (short)0, acc.c[0][0], false, false);
        acc.c[0][1] = __builtin_amdgcn_wmma_f32_16x16x32_bf16(
                          false, a0, false, b1, (short)0, acc.c[0][1], false, false);
        acc.c[1][0] = __builtin_amdgcn_wmma_f32_16x16x32_bf16(
                          false, a1, false, b0, (short)0, acc.c[1][0], false, false);
        acc.c[1][1] = __builtin_amdgcn_wmma_f32_16x16x32_bf16(
                          false, a1, false, b1, (short)0, acc.c[1][1], false, false);
    }
}

// ---------------------------------------------------------------- fp32 -> bf16 (x4/thread)
__global__ __launch_bounds__(256) void cvt4_bf16(
    const float* __restrict__ in, __bf16* __restrict__ out, int n4)
{
    int i = blockIdx.x * 256 + threadIdx.x;
    if (i < n4) {
        float4 f = reinterpret_cast<const float4*>(in)[i];
        uint2 o;
        o.x = pack2(f.x, f.y);
        o.y = pack2(f.z, f.w);
        reinterpret_cast<uint2*>(out)[i] = o;
    }
}

// ---------------------------------------------------------------- DMxDM transpose + cvt
__global__ __launch_bounds__(256) void transpose_cvt(
    const float* __restrict__ in, __bf16* __restrict__ out)
{
    __shared__ __bf16 tile[32][33];
    const int tx = threadIdx.x & 31;
    const int ty = threadIdx.x >> 5;           // 8 rows / pass
    const int bx = blockIdx.x * 32, by = blockIdx.y * 32;
#pragma unroll
    for (int r = 0; r < 4; ++r) {
        int y = ty + r * 8;
        tile[y][tx] = f32_bf(in[(size_t)(by + y) * DM + bx + tx]);
    }
    __syncthreads();
#pragma unroll
    for (int r = 0; r < 4; ++r) {
        int y = ty + r * 8;
        out[(size_t)(bx + y) * DM + by + tx] = tile[tx][y];
    }
}

// ---------------------------------------------------------------- projection
// z=0: Q (normal), z=1: K (normal), z=2: V stored transposed per batch (DM x SEQ)
__global__ __launch_bounds__(256) void proj_gemm(
    const __bf16* __restrict__ Xq, const __bf16* __restrict__ Xk,
    const __bf16* __restrict__ Xv,
    const __bf16* __restrict__ WT, const float* __restrict__ bias,
    __bf16* __restrict__ Oq, __bf16* __restrict__ Ok, __bf16* __restrict__ Vt)
{
    const __bf16* X = (blockIdx.z == 0) ? Xq : (blockIdx.z == 1 ? Xk : Xv);
    const int lane = threadIdx.x & 31;
    const int tile = blockIdx.x * 8 + (threadIdx.x >> 5);  // 8192 tiles
    const int row0 = (tile >> 5) * 32;                     // M=8192 -> 256
    const int col0 = (tile & 31) * 32;                     // N=1024 -> 32

    Acc2x2 acc{};
    gemm_tile_2x2(X, DM, WT, DM, row0, col0, DM, lane, acc);

    const int l15  = lane & 15;
    const int rsel = (lane < 16) ? 0 : 8;
    const float bb[2] = { bias[col0 + l15], bias[col0 + 16 + l15] };

    if (blockIdx.z < 2) {
        __bf16* O = (blockIdx.z == 0) ? Oq : Ok;
#pragma unroll
        for (int ti = 0; ti < 2; ++ti)
#pragma unroll
            for (int j = 0; j < 8; ++j) {
                const size_t r = row0 + ti * 16 + rsel + j;
                O[r * DM + col0 + l15]      = f32_bf(acc.c[ti][0][j] + bb[0]);
                O[r * DM + col0 + 16 + l15] = f32_bf(acc.c[ti][1][j] + bb[1]);
            }
    } else {
#pragma unroll
        for (int ti = 0; ti < 2; ++ti)
#pragma unroll
            for (int tj = 0; tj < 2; ++tj)
#pragma unroll
                for (int j = 0; j < 8; ++j) {
                    const int m = row0 + ti * 16 + rsel + j;   // b*SEQ + s
                    const int n = col0 + tj * 16 + l15;        // d
                    Vt[((size_t)(m >> 11) * DM + n) * SEQ + (m & (SEQ - 1))] =
                        f32_bf(acc.c[ti][tj][j] + bb[tj]);
                }
    }
}

// ---------------------------------------------------------------- scores = Q K^T (fp32 out)
__global__ __launch_bounds__(256) void score_gemm(
    const __bf16* __restrict__ Q, const __bf16* __restrict__ K,
    float* __restrict__ Sc)
{
    const int bt   = blockIdx.y;
    const int lane = threadIdx.x & 31;
    const int tile = blockIdx.x * 8 + (threadIdx.x >> 5);  // 4096 / batch
    const int row0 = (tile >> 6) * 32;
    const int col0 = (tile & 63) * 32;

    const __bf16* Qb = Q + (size_t)bt * SEQ * DM;
    const __bf16* Kb = K + (size_t)bt * SEQ * DM;   // already "transposed" for B use
    float*        Sb = Sc + (size_t)bt * SEQ * SEQ;

    Acc2x2 acc{};
    gemm_tile_2x2(Qb, DM, Kb, DM, row0, col0, DM, lane, acc);

    const int l15  = lane & 15;
    const int rsel = (lane < 16) ? 0 : 8;
#pragma unroll
    for (int ti = 0; ti < 2; ++ti)
#pragma unroll
        for (int j = 0; j < 8; ++j) {
            const size_t r = row0 + ti * 16 + rsel + j;
            Sb[r * SEQ + col0 + l15]      = acc.c[ti][0][j];
            Sb[r * SEQ + col0 + 16 + l15] = acc.c[ti][1][j];
        }
}

// ---------------------------------------------------------------- softmax (fp32 in place + bf16 copy)
__global__ __launch_bounds__(256) void softmax_rows(
    float* __restrict__ W, __bf16* __restrict__ Wb)
{
    __shared__ float red[256];
    const int tid = threadIdx.x;
    float*  p  = W  + (size_t)blockIdx.x * SEQ;
    __bf16* pb = Wb + (size_t)blockIdx.x * SEQ;

    float v[8];
    float m = -3.4e38f;
#pragma unroll
    for (int j = 0; j < 8; ++j) { v[j] = p[tid + 256 * j]; m = fmaxf(m, v[j]); }
    red[tid] = m; __syncthreads();
    for (int s = 128; s > 0; s >>= 1) {
        if (tid < s) red[tid] = fmaxf(red[tid], red[tid + s]);
        __syncthreads();
    }
    m = red[0]; __syncthreads();

    float sum = 0.f;
#pragma unroll
    for (int j = 0; j < 8; ++j) { v[j] = __expf(v[j] - m); sum += v[j]; }
    red[tid] = sum; __syncthreads();
    for (int s = 128; s > 0; s >>= 1) {
        if (tid < s) red[tid] += red[tid + s];
        __syncthreads();
    }
    const float inv = 1.f / red[0];
#pragma unroll
    for (int j = 0; j < 8; ++j) {
        const float o = v[j] * inv;
        p[tid + 256 * j]  = o;
        pb[tid + 256 * j] = f32_bf(o);
    }
}

// ---------------------------------------------------------------- ctx = weights @ V  (via V^T)
__global__ __launch_bounds__(256) void ctx_gemm(
    const __bf16* __restrict__ Wb, const __bf16* __restrict__ Vt,
    __bf16* __restrict__ C)
{
    const int bt   = blockIdx.y;
    const int lane = threadIdx.x & 31;
    const int tile = blockIdx.x * 8 + (threadIdx.x >> 5);  // 2048 / batch
    const int row0 = (tile >> 5) * 32;                     // S -> 64
    const int col0 = (tile & 31) * 32;                     // D -> 32

    const __bf16* Ab = Wb + (size_t)bt * SEQ * SEQ;
    const __bf16* Bb = Vt + (size_t)bt * DM * SEQ;
    __bf16*       Cb = C  + (size_t)bt * SEQ * DM;

    Acc2x2 acc{};
    gemm_tile_2x2(Ab, SEQ, Bb, SEQ, row0, col0, SEQ, lane, acc);

    const int l15  = lane & 15;
    const int rsel = (lane < 16) ? 0 : 8;
#pragma unroll
    for (int ti = 0; ti < 2; ++ti)
#pragma unroll
        for (int j = 0; j < 8; ++j) {
            const size_t r = row0 + ti * 16 + rsel + j;
            Cb[r * DM + col0 + l15]      = f32_bf(acc.c[ti][0][j]);
            Cb[r * DM + col0 + 16 + l15] = f32_bf(acc.c[ti][1][j]);
        }
}

// ---------------------------------------------------------------- out = ctx @ Wo + bo (fp32)
__global__ __launch_bounds__(256) void out_gemm(
    const __bf16* __restrict__ Cx, const __bf16* __restrict__ WoT,
    const float* __restrict__ bo, float* __restrict__ Out)
{
    const int lane = threadIdx.x & 31;
    const int tile = blockIdx.x * 8 + (threadIdx.x >> 5);  // 8192 tiles
    const int row0 = (tile >> 5) * 32;
    const int col0 = (tile & 31) * 32;

    Acc2x2 acc{};
    gemm_tile_2x2(Cx, DM, WoT, DM, row0, col0, DM, lane, acc);

    const int l15  = lane & 15;
    const int rsel = (lane < 16) ? 0 : 8;
    const float bb[2] = { bo[col0 + l15], bo[col0 + 16 + l15] };
#pragma unroll
    for (int ti = 0; ti < 2; ++ti)
#pragma unroll
        for (int j = 0; j < 8; ++j) {
            const size_t r = row0 + ti * 16 + rsel + j;
            Out[r * DM + col0 + l15]      = acc.c[ti][0][j] + bb[0];
            Out[r * DM + col0 + 16 + l15] = acc.c[ti][1][j] + bb[1];
        }
}

// ---------------------------------------------------------------- launch
extern "C" void kernel_launch(void* const* d_in, const int* in_sizes, int n_in,
                              void* d_out, int out_size, void* d_ws, size_t ws_size,
                              hipStream_t stream)
{
    (void)in_sizes; (void)n_in; (void)out_size; (void)ws_size;

    const float* query = (const float*)d_in[0];
    const float* key   = (const float*)d_in[1];
    const float* value = (const float*)d_in[2];
    const float* Wq    = (const float*)d_in[3];
    const float* bq    = (const float*)d_in[4];
    const float* Wo    = (const float*)d_in[5];
    const float* bo    = (const float*)d_in[6];

    float* out     = (float*)d_out;                       // (B,S,D)
    float* weights = out + (size_t)BAT * SEQ * DM;        // (B,S,T)

    const size_t NX = (size_t)BAT * SEQ * DM;             // 8M elems
    __bf16* WqT = (__bf16*)d_ws;                          // DM*DM
    __bf16* WoT = WqT + (size_t)DM * DM;
    __bf16* Xqb = WoT + (size_t)DM * DM;                  // inputs, bf16
    __bf16* Xkb = Xqb + NX;
    __bf16* Xvb = Xkb + NX;
    __bf16* Qb  = Xvb + NX;                               // projected
    __bf16* Kb  = Qb + NX;
    __bf16* Vt  = Kb + NX;                                // V transposed (B, D, S)
    __bf16* Cb  = Vt + NX;                                // ctx
    __bf16* Pb  = Cb + NX;                                // softmax weights bf16 (B*S*T)

    const int n4 = (int)(NX / 4);
    cvt4_bf16<<<n4 / 256, 256, 0, stream>>>(query, Xqb, n4);
    cvt4_bf16<<<n4 / 256, 256, 0, stream>>>(key,   Xkb, n4);
    cvt4_bf16<<<n4 / 256, 256, 0, stream>>>(value, Xvb, n4);
    transpose_cvt<<<dim3(DM / 32, DM / 32), 256, 0, stream>>>(Wq, WqT);
    transpose_cvt<<<dim3(DM / 32, DM / 32), 256, 0, stream>>>(Wo, WoT);

    proj_gemm<<<dim3(1024, 1, 3), 256, 0, stream>>>(Xqb, Xkb, Xvb, WqT, bq,
                                                    Qb, Kb, Vt);
    score_gemm<<<dim3(512, BAT), 256, 0, stream>>>(Qb, Kb, weights);
    softmax_rows<<<BAT * SEQ, 256, 0, stream>>>(weights, Pb);
    ctx_gemm<<<dim3(256, BAT), 256, 0, stream>>>(Pb, Vt, Cb);
    out_gemm<<<1024, 256, 0, stream>>>(Cb, WoT, bo, out);
}
